// LSTM1_40939628266119
// MI455X (gfx1250) — matrix operand order repaired and checked by reading, not compiled, not run on hardware
//
#include <hip/hip_runtime.h>

typedef __attribute__((ext_vector_type(16))) _Float16 v16h;
typedef __attribute__((ext_vector_type(8)))  float    v8f;

#define B_   64
#define T_   800
#define U_   64
#define V_   60
#define C_   512
#define K_   10
#define NWG  32
#define THR  128
#define GC   16      // h-columns owned per WG

union Frag { v16h v; uint4 q[2]; };

__device__ __forceinline__ float sig_(float x) { return 1.0f / (1.0f + __expf(-x)); }

// ---------------- prep: f32->f16 weights, bias fuse, onehot->ids, h0, barrier reset ----
__global__ void prep_kernel(const float* __restrict__ Wih, const float* __restrict__ Whh,
                            const float* __restrict__ winW, const float* __restrict__ bih,
                            const float* __restrict__ bhh, const float* __restrict__ onehots,
                            const float* __restrict__ prev_h,
                            _Float16* Wih16, _Float16* Whh16, _Float16* WinW16,
                            float* bias, unsigned char* ids, _Float16* hbuf1, unsigned* bar) {
  int i0 = blockIdx.x * blockDim.x + threadIdx.x;
  int st = gridDim.x * blockDim.x;
  for (int i = i0; i < 2048 * 512; i += st) Whh16[i] = (_Float16)Whh[i];
  for (int i = i0; i < 2048 * 64; i += st) {
    int n = i >> 6, k = i & 63;
    Wih16[i] = (_Float16)((k < 63) ? Wih[n * 63 + k] : 0.0f);
  }
  for (int i = i0; i < 32 * 512; i += st) WinW16[i] = (_Float16)((i < 30 * 512) ? winW[i] : 0.0f);
  for (int i = i0; i < 2048; i += st) bias[i] = bih[i] + bhh[i];
  for (int i = i0; i < 64 * 64; i += st) {
    const float* row = onehots + (size_t)i * 60;
    unsigned char id = 0;
    for (int v = 0; v < 60; ++v) if (row[v] > 0.5f) id = (unsigned char)v;
    ids[i] = id;
  }
  for (int i = i0; i < 64 * 512; i += st) hbuf1[i] = (_Float16)prev_h[i];
  if (i0 < 8) bar[i0] = 0u;
}

// ---------------- persistent LSTM + attention kernel --------------------------------
__global__ __launch_bounds__(THR, 1)
void lstm_attn_kernel(const float* __restrict__ x, const float* __restrict__ text_lens,
                      const float* __restrict__ w_old, const float* __restrict__ kappa_old,
                      const float* __restrict__ prev_c, const float* __restrict__ win_b,
                      const float* __restrict__ bias,
                      const _Float16* __restrict__ Whh16, const _Float16* __restrict__ Wih16,
                      const _Float16* __restrict__ WinW16, const unsigned char* __restrict__ idsg,
                      _Float16* hbuf, unsigned* bar,
                      float* o_ws, float* o_hs, float* o_h, float* o_c,
                      float* o_w, float* o_kap, float* o_phi) {
  __shared__ __align__(16) _Float16 sWhh[64 * 512];   // 64 KB  weight slice (i,f,g,o x 16 cols)
  __shared__ __align__(16) _Float16 sWih[64 * 64];    //  8 KB
  __shared__ __align__(16) _Float16 sWinW[32 * 512];  // 32 KB
  __shared__ __align__(16) _Float16 sInp[64 * 64];    //  8 KB  [x(3), w(60), 0]
  __shared__ __align__(16) float    sP[64 * 32];      //  8 KB  exp(h@winW.T + b)
  __shared__ __align__(16) float    sPhi[64 * 65];
  __shared__ __align__(16) float    sKap[64 * 10];
  __shared__ __align__(16) float    sW[64 * 60];
  __shared__ float sScale[64];
  __shared__ float sWinB[32];
  __shared__ __align__(4) unsigned char sIds[64 * 64];

  const int wg   = blockIdx.x;
  const int tid  = threadIdx.x;
  const int lane = tid & 31;
  const int wv   = tid >> 5;          // wave id == M-tile (batch tile)
  const int ln15 = lane & 15;
  const int hi   = (lane >> 4) & 1;
  const int selA = hi ? 8 : 0;        // A-frag K sub-select (16-bit A layout)
  const int selB = hi ? 8 : 0;        // B-frag K sub-select (16-bit B layout)
  const int col0 = wg * GC;
  const int brow0 = wv * 16;

  // ---- stage weights / state into LDS ----
  for (int i = tid; i < 64 * 64; i += THR) {          // Whh slice: 64 rows x 64 uint4
    int nl = i >> 6, ch = i & 63;
    int src = ((nl >> 4) * 512 + wg * GC + (nl & 15));
    ((uint4*)sWhh)[i] = ((const uint4*)Whh16)[src * 64 + ch];
  }
  for (int i = tid; i < 64 * 8; i += THR) {           // Wih slice: 64 rows x 8 uint4
    int nl = i >> 3, ch = i & 7;
    int src = ((nl >> 4) * 512 + wg * GC + (nl & 15));
    ((uint4*)sWih)[i] = ((const uint4*)Wih16)[src * 8 + ch];
  }
  for (int i = tid; i < 32 * 64; i += THR) ((uint4*)sWinW)[i] = ((const uint4*)WinW16)[i];
  for (int i = tid; i < 1024; i += THR) ((unsigned*)sIds)[i] = ((const unsigned*)idsg)[i];
  for (int i = tid; i < 64 * 60; i += THR) sW[i] = w_old[i];
  for (int i = tid; i < 640; i += THR) sKap[i] = kappa_old[i];
  if (tid < 64) sScale[tid] = (float)U_ / text_lens[tid];
  if (tid < 32) sWinB[tid] = (tid < 30) ? win_b[tid] : 0.0f;

  // c state lives in registers in accumulator layout
  float cst[8];
  for (int r = 0; r < 8; ++r)
    cst[r] = prev_c[(brow0 + r + hi * 8) * C_ + col0 + ln15];
  const float bi_ = bias[0 * 512 + col0 + ln15];
  const float bf_ = bias[1 * 512 + col0 + ln15];
  const float bg_ = bias[2 * 512 + col0 + ln15];
  const float bo_ = bias[3 * 512 + col0 + ln15];
  __syncthreads();

  const v8f vzero = {0.f, 0.f, 0.f, 0.f, 0.f, 0.f, 0.f, 0.f};
  unsigned* cnt = bar;
  unsigned* gen = bar + 1;

  for (int t = 0; t < T_; ++t) {
    // ---- phase A: build inp = [x_t, w, 0] in f16 ----
    for (int i = tid; i < 64 * 64; i += THR) {
      int b = i >> 6, j = i & 63;
      float v = (j < 3) ? x[(b * T_ + t) * 3 + j] : ((j < 63) ? sW[b * 60 + (j - 3)] : 0.0f);
      sInp[i] = (_Float16)v;
    }
    __syncthreads();

    // ---- phase B: gates via WMMA (K = 64 inp + 512 h) ----
    const _Float16* hprev = hbuf + ((t & 1) ^ 1) * (B_ * C_);
    v8f acc[4] = {vzero, vzero, vzero, vzero};
    for (int ks = 0; ks < 2; ++ks) {
      int kb = ks * 32;
      Frag a;
      const _Float16* ap = &sInp[(brow0 + ln15) * 64 + kb + selA];
      a.q[0] = *(const uint4*)ap;  a.q[1] = *(const uint4*)(ap + 16);
      for (int g = 0; g < 4; ++g) {
        Frag b;
        const _Float16* bp = &sWih[(g * 16 + ln15) * 64 + kb + selB];
        b.q[0] = *(const uint4*)bp;  b.q[1] = *(const uint4*)(bp + 16);
        acc[g] = __builtin_amdgcn_wmma_f32_16x16x32_f16(false, a.v, false, b.v,
                                                        (short)0, acc[g], false, false);
      }
    }
    for (int ks = 0; ks < 16; ++ks) {
      int kb = ks * 32;
      Frag a;
      const _Float16* ap = &hprev[(brow0 + ln15) * C_ + kb + selA];
      a.q[0] = *(const uint4*)ap;  a.q[1] = *(const uint4*)(ap + 16);
      for (int g = 0; g < 4; ++g) {
        Frag b;
        const _Float16* bp = &sWhh[(g * 16 + ln15) * C_ + kb + selB];
        b.q[0] = *(const uint4*)bp;  b.q[1] = *(const uint4*)(bp + 16);
        acc[g] = __builtin_amdgcn_wmma_f32_16x16x32_f16(false, a.v, false, b.v,
                                                        (short)0, acc[g], false, false);
      }
    }

    // ---- LSTM pointwise; c in registers; emit h (f16 exchange + f32 hs output) ----
    _Float16* hcur = hbuf + (t & 1) * (B_ * C_);
    for (int r = 0; r < 8; ++r) {
      int b = brow0 + r + hi * 8;
      float iv = sig_(acc[0][r] + bi_);
      float fv = sig_(acc[1][r] + bf_);
      float gv = tanhf(acc[2][r] + bg_);
      float ov = sig_(acc[3][r] + bo_);
      float cn = fv * cst[r] + iv * gv;
      cst[r] = cn;
      float hv = ov * tanhf(cn);
      hcur[b * C_ + col0 + ln15] = (_Float16)hv;
      o_hs[(size_t)b * T_ * C_ + (size_t)t * C_ + col0 + ln15] = hv;
      if (t == T_ - 1) {
        o_h[b * C_ + col0 + ln15] = hv;
        o_c[b * C_ + col0 + ln15] = cn;
      }
    }

    // ---- phase C: device-wide sense barrier (h(t) published) ----
    __threadfence();
    __syncthreads();
    if (tid == 0) {
      unsigned g = __hip_atomic_load(gen, __ATOMIC_ACQUIRE, __HIP_MEMORY_SCOPE_AGENT);
      unsigned a = __hip_atomic_fetch_add(cnt, 1u, __ATOMIC_ACQ_REL, __HIP_MEMORY_SCOPE_AGENT);
      if (a == NWG - 1) {
        __hip_atomic_store(cnt, 0u, __ATOMIC_RELAXED, __HIP_MEMORY_SCOPE_AGENT);
        __hip_atomic_fetch_add(gen, 1u, __ATOMIC_ACQ_REL, __HIP_MEMORY_SCOPE_AGENT);
      } else {
        while (__hip_atomic_load(gen, __ATOMIC_ACQUIRE, __HIP_MEMORY_SCOPE_AGENT) == g)
          __builtin_amdgcn_s_sleep(1);
      }
    }
    __syncthreads();
    __threadfence();

    // ---- phase D: attention window (redundant per WG): p = exp(h @ winW.T + b) ----
    v8f pacc[2] = {vzero, vzero};
    for (int ks = 0; ks < 16; ++ks) {
      int kb = ks * 32;
      Frag a;
      const _Float16* ap = &hcur[(brow0 + ln15) * C_ + kb + selA];
      a.q[0] = *(const uint4*)ap;  a.q[1] = *(const uint4*)(ap + 16);
      for (int nt = 0; nt < 2; ++nt) {
        Frag b;
        const _Float16* bp = &sWinW[(nt * 16 + ln15) * C_ + kb + selB];
        b.q[0] = *(const uint4*)bp;  b.q[1] = *(const uint4*)(bp + 16);
        pacc[nt] = __builtin_amdgcn_wmma_f32_16x16x32_f16(false, a.v, false, b.v,
                                                          (short)0, pacc[nt], false, false);
      }
    }
    for (int nt = 0; nt < 2; ++nt) {
      int n = nt * 16 + ln15;
      float wb = sWinB[n];
      for (int r = 0; r < 8; ++r)
        sP[(brow0 + r + hi * 8) * 32 + n] = __expf(pacc[nt][r] + wb);
    }
    __syncthreads();

    if (tid < 64)                                   // kappa += dkappa
      for (int k = 0; k < 10; ++k) sKap[tid * 10 + k] += sP[tid * 32 + 20 + k];
    __syncthreads();

    {                                               // phi: 2 threads per batch row
      int b = tid >> 1, half = tid & 1;
      float al[10], be[10], kp[10];
      for (int k = 0; k < 10; ++k) {
        al[k] = sP[b * 32 + k]; be[k] = sP[b * 32 + 10 + k]; kp[k] = sKap[b * 10 + k];
      }
      float sc = sScale[b];
      int u0 = half ? 33 : 0, u1 = half ? 65 : 33;
      for (int u = u0; u < u1; ++u) {
        float s = 0.f;
        for (int k = 0; k < 10; ++k) { float d = kp[k] - (float)u; s += al[k] * __expf(-be[k] * d * d); }
        sPhi[b * 65 + u] = s * sc;
      }
    }
    __syncthreads();

    if (tid < 64) {                                 // w = phi . onehot  (scatter via char ids)
      int b = tid;
      for (int v = 0; v < 60; ++v) sW[b * 60 + v] = 0.f;
      for (int u = 0; u < 64; ++u) sW[b * 60 + sIds[b * 64 + u]] += sPhi[b * 65 + u];
    }
    __syncthreads();

    if (wg == 0) {
      for (int i = tid; i < 64 * 60; i += THR) {
        int b = i / 60, v = i - b * 60;
        o_ws[(size_t)b * T_ * V_ + (size_t)t * V_ + v] = sW[i];
      }
      if (t == T_ - 1) {
        for (int i = tid; i < 64 * 60; i += THR) o_w[i] = sW[i];
        for (int i = tid; i < 640; i += THR) o_kap[i] = sKap[i];
        for (int i = tid; i < 64 * 65; i += THR) o_phi[i] = sPhi[i];
      }
    }
    __syncthreads();
  }
}

extern "C" void kernel_launch(void* const* d_in, const int* in_sizes, int n_in,
                              void* d_out, int out_size, void* d_ws, size_t ws_size,
                              hipStream_t stream) {
  const float* x         = (const float*)d_in[0];
  const float* onehots   = (const float*)d_in[1];
  const float* text_lens = (const float*)d_in[2];
  const float* w_old     = (const float*)d_in[3];
  const float* kappa_old = (const float*)d_in[4];
  const float* prev_h    = (const float*)d_in[5];
  const float* prev_c    = (const float*)d_in[6];
  const float* W_ih      = (const float*)d_in[7];
  const float* W_hh      = (const float*)d_in[8];
  const float* b_ih      = (const float*)d_in[9];
  const float* b_hh      = (const float*)d_in[10];
  const float* win_W     = (const float*)d_in[11];
  const float* win_b     = (const float*)d_in[12];

  char* ws = (char*)d_ws;
  _Float16* Whh16  = (_Float16*)ws;       ws += 2048 * 512 * 2;
  _Float16* Wih16  = (_Float16*)ws;       ws += 2048 * 64 * 2;
  _Float16* WinW16 = (_Float16*)ws;       ws += 32 * 512 * 2;
  float*    bias   = (float*)ws;          ws += 2048 * 4;
  unsigned char* ids = (unsigned char*)ws; ws += 64 * 64;
  _Float16* hbuf   = (_Float16*)ws;       ws += 2 * 64 * 512 * 2;
  unsigned* bar    = (unsigned*)ws;       ws += 256;

  float* o = (float*)d_out;
  float* o_ws_  = o;                 // [B,T,V]
  float* o_hs_  = o + 3072000;       // [B,T,C]
  float* o_h_   = o + 29286400;      // [B,C]
  float* o_c_   = o + 29319168;      // [B,C]
  float* o_w_   = o + 29351936;      // [B,V]
  float* o_kap_ = o + 29355776;      // [B,K]
  float* o_phi_ = o + 29356416;      // [B,U+1]

  prep_kernel<<<4096, 256, 0, stream>>>(W_ih, W_hh, win_W, b_ih, b_hh, onehots, prev_h,
                                        Wih16, Whh16, WinW16, bias, ids,
                                        hbuf + 64 * 512, bar);
  lstm_attn_kernel<<<NWG, THR, 0, stream>>>(x, text_lens, w_old, kappa_old, prev_c, win_b,
                                            bias, Whh16, Wih16, WinW16, ids, hbuf, bar,
                                            o_ws_, o_hs_, o_h_, o_c_, o_w_, o_kap_, o_phi_);
}
